// StackedLSTM_17944373363097
// MI455X (gfx1250) — matrix operand order repaired and checked by reading, not compiled, not run on hardware
//
#include <hip/hip_runtime.h>
#include <hip/hip_bf16.h>
#include <math.h>

// Problem constants (from reference)
#define B_   512
#define T_   2048
#define H_   128
#define G4H  512   // 4*H
#define NC_  6

#define MT   16    // batch-tile rows per block
#define NTPB 512   // 16 wave32s per block
#define HS   136   // h/x LDS row stride (halves)
#define GS   516   // gates LDS row stride (floats)

typedef __attribute__((ext_vector_type(16))) _Float16 v16h;
typedef __attribute__((ext_vector_type(8)))  float    v8f;
typedef __attribute__((ext_vector_type(4)))  _Float16 v4h;

#define LOG2E 1.44269504088896340736f

// Branch-free fast transcendentals on the native v_exp_f32 / v_rcp_f32 pipes.
__device__ __forceinline__ float fast_exp(float x) {
    return __builtin_amdgcn_exp2f(x * LOG2E);
}
__device__ __forceinline__ float sigf(float x) {
    return __builtin_amdgcn_rcpf(1.0f + fast_exp(-x));
}
__device__ __forceinline__ float tanh_fast(float x) {
    const float ax = __builtin_fabsf(x);
    const float e  = fast_exp(-2.0f * ax);
    const float r  = (1.0f - e) * __builtin_amdgcn_rcpf(1.0f + e);
    return __builtin_copysignf(r, x);
}

// Build a 32x16 (KxN) f16 B-fragment for wmma_f32_16x16x32_f16 from a row-major
// [N][K]=[512][128] f32 weight matrix (B[k][n] = W[n][k]).
// Per ISA 7.12.2: lane col = lane&15; lanes 0-15 hold K={k0..k0+7, k0+16..k0+23}, lanes 16-31 K={k0+8..k0+15, k0+24..k0+31}.
__device__ __forceinline__ v16h make_bfrag(const float* __restrict__ W, int n, int k0, int hi) {
    const float* p = W + n * H_ + k0 + hi * 8;
    v16h b;
#pragma unroll
    for (int q = 0; q < 16; ++q) b[q] = (_Float16)p[q < 8 ? q : q + 8];
    return b;
}

// Build a 16x32 (MxK) f16 A-fragment from an LDS row-major [16][HS] f16 buffer.
// Per ISA 7.12.2: row = lane&15; lanes 0-15: halves = K {k0..k0+7, k0+16..k0+23};
// lanes 16-31: halves = K {k0+8..k0+15, k0+24..k0+31}.
__device__ __forceinline__ v16h make_afrag(const _Float16* __restrict__ buf, int lane, int k0) {
    const int row = lane & 15;
    const int hi  = lane >> 4;
    const _Float16* p = buf + row * HS + k0 + hi * 8;
    v16h a;
#pragma unroll
    for (int q = 0; q < 8; ++q) { a[q] = p[q]; a[8 + q] = p[16 + q]; }
    return a;
}

// ---------------- Layer 0: feats -> LSTM(2->128), writes y0 [B,T,H] f16 ----------------
__global__ __launch_bounds__(NTPB)
void lstm0_kernel(const float* __restrict__ x,
                  const float* __restrict__ Wih0,
                  const float* __restrict__ Whh0,
                  const float* __restrict__ bih0,
                  const float* __restrict__ bhh0,
                  _Float16* __restrict__ y0)
{
    __shared__ _Float16 hlds[MT * HS];
    __shared__ float    gates[MT * GS];
    __shared__ float    fa_s[MT], fg_s[MT];

    const int tid  = threadIdx.x;
    const int lane = tid & 31;
    const int wv   = tid >> 5;          // 0..15
    const int col  = lane & 15;
    const int hi   = lane >> 4;
    const int rb   = blockIdx.x * MT;

    // Recurrent weights as persistent register B-fragments: wave covers 2 N-tiles.
    v16h bfh[2][4];
    float w0c[2], w1c[2], bsc[2];
#pragma unroll
    for (int j = 0; j < 2; ++j) {
        const int n = (wv * 2 + j) * 16 + col;
        w0c[j] = Wih0[n * 2 + 0];
        w1c[j] = Wih0[n * 2 + 1];
        bsc[j] = bih0[n] + bhh0[n];
#pragma unroll
        for (int kc = 0; kc < 4; ++kc) bfh[j][kc] = make_bfrag(Whh0, n, kc * 32, hi);
    }

    for (int i = tid; i < MT * HS; i += NTPB) hlds[i] = (_Float16)0.0f;

    // Elementwise ownership: wave w owns row w; lane owns 4 consecutive hidden units.
    const int erow = (tid * 4) >> 7;
    const int ej0  = (tid * 4) & 127;
    float cc[4] = {0.f, 0.f, 0.f, 0.f};

    __syncthreads();

    for (int t = 0; t < T_; ++t) {
        // ---- stage: accel/gyro norms for this timestep ----
        if (tid < MT) {
            const float* xp = x + ((size_t)(rb + tid) * T_ + t) * 6;
            float a0 = xp[0], a1 = xp[1], a2 = xp[2];
            float g0 = xp[3], g1 = xp[4], g2 = xp[5];
            fa_s[tid] = sqrtf(a0 * a0 + a1 * a1 + a2 * a2);
            fg_s[tid] = sqrtf(g0 * g0 + g1 * g1 + g2 * g2);
            if (t + 1 < T_) __builtin_prefetch(xp + 6, 0, 1);
        }
        __syncthreads();

        // ---- gates = xg0 + h @ Whh0^T via WMMA ----
        // Preload all A-fragments so the ds_loads pipeline ahead of the WMMA burst.
        v16h afr[4];
#pragma unroll
        for (int kc = 0; kc < 4; ++kc) afr[kc] = make_afrag(hlds, lane, kc * 32);

        float fa8[8], fg8[8];
#pragma unroll
        for (int r = 0; r < 8; ++r) { fa8[r] = fa_s[r + 8 * hi]; fg8[r] = fg_s[r + 8 * hi]; }

        v8f acc[2];
#pragma unroll
        for (int j = 0; j < 2; ++j)
#pragma unroll
            for (int r = 0; r < 8; ++r)
                acc[j][r] = fa8[r] * w0c[j] + fg8[r] * w1c[j] + bsc[j];

#pragma unroll
        for (int kc = 0; kc < 4; ++kc)
#pragma unroll
            for (int j = 0; j < 2; ++j)
                acc[j] = __builtin_amdgcn_wmma_f32_16x16x32_f16(
                    false, afr[kc], false, bfh[j][kc], (short)0, acc[j], false, false);

#pragma unroll
        for (int j = 0; j < 2; ++j) {
            const int n = (wv * 2 + j) * 16 + col;
#pragma unroll
            for (int r = 0; r < 8; ++r) gates[(r + 8 * hi) * GS + n] = acc[j][r];
        }
        __syncthreads();

        // ---- elementwise LSTM cell update (branch-free transcendentals) ----
        v4h hv;
#pragma unroll
        for (int e = 0; e < 4; ++e) {
            const int j = ej0 + e;
            const float gi = gates[erow * GS + j];
            const float gf = gates[erow * GS + 128 + j];
            const float gg = gates[erow * GS + 256 + j];
            const float go = gates[erow * GS + 384 + j];
            const float cn = sigf(gf) * cc[e] + sigf(gi) * tanh_fast(gg);
            cc[e] = cn;
            const float hh = sigf(go) * tanh_fast(cn);
            hlds[erow * HS + j] = (_Float16)hh;
            hv[e] = (_Float16)hh;
        }
        *(v4h*)(y0 + ((size_t)(rb + erow) * T_ + t) * H_ + ej0) = hv;
    }
}

// ---- Layer 1: LSTM(128->128) with fused input GEMM; writes final h [B,H] f32 ----
__global__ __launch_bounds__(NTPB)
void lstm1_kernel(const _Float16* __restrict__ y0,
                  const float* __restrict__ Wih1,
                  const float* __restrict__ Whh1,
                  const float* __restrict__ bih1,
                  const float* __restrict__ bhh1,
                  float* __restrict__ h1out)
{
    __shared__ _Float16 hlds[MT * HS];
    __shared__ _Float16 xlds[MT * HS];
    __shared__ float    gates[MT * GS];

    const int tid  = threadIdx.x;
    const int lane = tid & 31;
    const int wv   = tid >> 5;
    const int col  = lane & 15;
    const int hi   = lane >> 4;
    const int rb   = blockIdx.x * MT;

    // Both weight matrices as persistent register B-fragments.
    v16h bfh[2][4], bfx[2][4];
    float bsc[2];
#pragma unroll
    for (int j = 0; j < 2; ++j) {
        const int n = (wv * 2 + j) * 16 + col;
        bsc[j] = bih1[n] + bhh1[n];
#pragma unroll
        for (int kc = 0; kc < 4; ++kc) {
            bfh[j][kc] = make_bfrag(Whh1, n, kc * 32, hi);
            bfx[j][kc] = make_bfrag(Wih1, n, kc * 32, hi);
        }
    }

    for (int i = tid; i < MT * HS; i += NTPB) hlds[i] = (_Float16)0.0f;

    const int erow = (tid * 4) >> 7;
    const int ej0  = (tid * 4) & 127;
    float cc[4] = {0.f, 0.f, 0.f, 0.f};

    __syncthreads();

    for (int t = 0; t < T_; ++t) {
        // ---- stage: y0[:, t, :] tile into LDS (one b64 per thread) ----
        {
            const int flat = tid * 4;
            const int row  = flat >> 7;
            const int off  = flat & 127;
            const _Float16* src = y0 + ((size_t)(rb + row) * T_ + t) * H_ + off;
            *(v4h*)&xlds[row * HS + off] = *(const v4h*)src;
            if (t + 1 < T_) __builtin_prefetch(src + H_, 0, 1);
        }
        __syncthreads();

        // ---- gates = bias + x_t @ Wih1^T + h @ Whh1^T via WMMA ----
        // Preload all 8 A-fragments (x_t and h) before the WMMA burst.
        v16h axr[4], ahr[4];
#pragma unroll
        for (int kc = 0; kc < 4; ++kc) {
            axr[kc] = make_afrag(xlds, lane, kc * 32);
            ahr[kc] = make_afrag(hlds, lane, kc * 32);
        }

        v8f acc[2];
#pragma unroll
        for (int j = 0; j < 2; ++j)
#pragma unroll
            for (int r = 0; r < 8; ++r) acc[j][r] = bsc[j];

#pragma unroll
        for (int kc = 0; kc < 4; ++kc)
#pragma unroll
            for (int j = 0; j < 2; ++j)
                acc[j] = __builtin_amdgcn_wmma_f32_16x16x32_f16(
                    false, axr[kc], false, bfx[j][kc], (short)0, acc[j], false, false);
#pragma unroll
        for (int kc = 0; kc < 4; ++kc)
#pragma unroll
            for (int j = 0; j < 2; ++j)
                acc[j] = __builtin_amdgcn_wmma_f32_16x16x32_f16(
                    false, ahr[kc], false, bfh[j][kc], (short)0, acc[j], false, false);

#pragma unroll
        for (int j = 0; j < 2; ++j) {
            const int n = (wv * 2 + j) * 16 + col;
#pragma unroll
            for (int r = 0; r < 8; ++r) gates[(r + 8 * hi) * GS + n] = acc[j][r];
        }
        __syncthreads();

        // ---- elementwise LSTM cell update ----
#pragma unroll
        for (int e = 0; e < 4; ++e) {
            const int j = ej0 + e;
            const float gi = gates[erow * GS + j];
            const float gf = gates[erow * GS + 128 + j];
            const float gg = gates[erow * GS + 256 + j];
            const float go = gates[erow * GS + 384 + j];
            const float cn = sigf(gf) * cc[e] + sigf(gi) * tanh_fast(gg);
            cc[e] = cn;
            const float hh = sigf(go) * tanh_fast(cn);
            hlds[erow * HS + j] = (_Float16)hh;
            if (t == T_ - 1) h1out[(size_t)(rb + erow) * H_ + j] = hh;
        }
    }
}

// ---------------- Classifier MLP 128->128->64->6 + double softmax ----------------
__global__ __launch_bounds__(64)
void mlp_kernel(const float* __restrict__ h1,
                const float* __restrict__ W1, const float* __restrict__ b1,
                const float* __restrict__ W2, const float* __restrict__ b2,
                const float* __restrict__ W3, const float* __restrict__ b3,
                float* __restrict__ out)
{
    __shared__ float hr[128], z1[128], z2[64], z3[8];
    const int tid = threadIdx.x;
    const int b   = blockIdx.x;

    hr[tid]      = h1[(size_t)b * H_ + tid];
    hr[tid + 64] = h1[(size_t)b * H_ + 64 + tid];
    __syncthreads();

#pragma unroll
    for (int rep = 0; rep < 2; ++rep) {
        const int o = tid + rep * 64;
        float s = b1[o];
        for (int k = 0; k < 128; ++k) s += W1[o * 128 + k] * hr[k];
        z1[o] = fmaxf(s, 0.0f);
    }
    __syncthreads();

    {
        float s = b2[tid];
        for (int k = 0; k < 128; ++k) s += W2[tid * 128 + k] * z1[k];
        z2[tid] = fmaxf(s, 0.0f);
    }
    __syncthreads();

    if (tid < NC_) {
        float s = b3[tid];
        for (int k = 0; k < 64; ++k) s += W3[tid * 64 + k] * z2[k];
        z3[tid] = s;
    }
    __syncthreads();

    if (tid == 0) {
        float v[NC_];
        for (int i = 0; i < NC_; ++i) v[i] = z3[i];
#pragma unroll
        for (int pass = 0; pass < 2; ++pass) {
            float m = v[0];
            for (int i = 1; i < NC_; ++i) m = fmaxf(m, v[i]);
            float sm = 0.0f;
            for (int i = 0; i < NC_; ++i) { v[i] = fast_exp(v[i] - m); sm += v[i]; }
            const float inv = 1.0f / sm;
            for (int i = 0; i < NC_; ++i) v[i] *= inv;
        }
        for (int i = 0; i < NC_; ++i) out[(size_t)b * NC_ + i] = v[i];
    }
}

extern "C" void kernel_launch(void* const* d_in, const int* in_sizes, int n_in,
                              void* d_out, int out_size, void* d_ws, size_t ws_size,
                              hipStream_t stream)
{
    const float* x    = (const float*)d_in[0];
    const float* Wih0 = (const float*)d_in[1];
    const float* Whh0 = (const float*)d_in[2];
    const float* bih0 = (const float*)d_in[3];
    const float* bhh0 = (const float*)d_in[4];
    const float* Wih1 = (const float*)d_in[5];
    const float* Whh1 = (const float*)d_in[6];
    const float* bih1 = (const float*)d_in[7];
    const float* bhh1 = (const float*)d_in[8];
    const float* W1   = (const float*)d_in[9];
    const float* b1   = (const float*)d_in[10];
    const float* W2   = (const float*)d_in[11];
    const float* b2   = (const float*)d_in[12];
    const float* W3   = (const float*)d_in[13];
    const float* b3   = (const float*)d_in[14];
    float* out = (float*)d_out;

    _Float16* y0 = (_Float16*)d_ws;                                   // [B,T,H] f16
    float*    h1 = (float*)((char*)d_ws + (size_t)B_ * T_ * H_ * 2);  // [B,H]  f32

    lstm0_kernel<<<B_ / MT, NTPB, 0, stream>>>(x, Wih0, Whh0, bih0, bhh0, y0);
    lstm1_kernel<<<B_ / MT, NTPB, 0, stream>>>(y0, Wih1, Whh1, bih1, bhh1, h1);
    mlp_kernel<<<B_, 64, 0, stream>>>(h1, W1, b1, W2, b2, W3, b3, out);
}
